// MHAModule_4466765988502
// MI455X (gfx1250) — matrix operand order repaired and checked by reading, not compiled, not run on hardware
//
#include <hip/hip_runtime.h>
#include <hip/hip_bf16.h>
#include <math.h>

// ---------------------------------------------------------------------------
// MI455X (gfx1250) implementation.
//   - Dense GEMMs use v_wmma_f32_16x16x32_bf16 (wave32, 16x16 tiles),
//     double-buffered LDS tiles, one barrier per K-step.
//   - The dominant-cost output 1x1 conv fuses bilinear x8 upsample + hardswish
//     into its B-operand loader so the 205MB intermediate never hits HBM.
// Shapes: B=32, C=Co=512, H=8, d=64, R=56, N=49 (7x7 grid), 32 GN groups.
// ---------------------------------------------------------------------------

typedef __attribute__((ext_vector_type(16))) __bf16 v16bf;
typedef __attribute__((ext_vector_type(8)))  float  v8f;

#define BB   32
#define CC   512
#define RR   56
#define NPIX 49
#define EPSN 1e-5f

// WMMA fragment layouts (CDNA5 ISA 7.12.2, wave32):
//  A (16x32 bf16): half=lane/16, m=lane%16;
//    VGPR v<4: K=2v+8h+bit ; v>=4: K=16+2(v-4)+8h+bit
//  B (32x16 bf16, LDS stored n-major): n=lane%16, K=a_frag_k(e,h) (same as A)
//  C/D (16x16 f32): M = 8h + r, N = lane%16
__device__ __forceinline__ int a_frag_k(int e, int half) {
    int vg = e >> 1;
    return (vg < 4 ? 2 * vg : 16 + 2 * (vg - 4)) + half * 8 + (e & 1);
}

// ---------------------------------------------------------------------------
// 1) depthwise 3x3 stride-8 conv (pad 1) + bias + eval-BN  -> [B,512,49]
// ---------------------------------------------------------------------------
__global__ __launch_bounds__(64)
void dwconv_bn_kernel(const float* __restrict__ X,   // [B,512,56,56]
                      const float* __restrict__ W,   // [512,1,3,3]
                      const float* __restrict__ bias,
                      const float* __restrict__ bng,
                      const float* __restrict__ bnb,
                      float* __restrict__ Y)         // [B,512,49]
{
    int bc = blockIdx.x;            // b*512 + c
    int c  = bc & (CC - 1);
    int n  = threadIdx.x;
    if (n >= NPIX) return;
    int oy = n / 7, ox = n % 7;
    const float* xin = X + (size_t)bc * (RR * RR);
    const float* w   = W + c * 9;
    float acc = 0.f;
#pragma unroll
    for (int r = 0; r < 3; ++r) {
        int iy = oy * 8 - 1 + r;
        if ((unsigned)iy >= (unsigned)RR) continue;
#pragma unroll
        for (int s = 0; s < 3; ++s) {
            int ix = ox * 8 - 1 + s;
            if ((unsigned)ix >= (unsigned)RR) continue;
            acc += w[r * 3 + s] * xin[iy * RR + ix];
        }
    }
    float scale = bng[c] * rsqrtf(1.f + EPSN);
    Y[(size_t)bc * NPIX + n] = (acc + bias[c]) * scale + bnb[c];
}

// ---------------------------------------------------------------------------
// 2) projection GEMM: Y[b,m,n] = sum_k W[m,k] * X[b,k,n]   (M=K=512, N=49)
//    Block: 256 thr (8 waves) = 256 M rows x 64 N cols (N zero-padded).
//    Each wave: 2 M-tiles x 4 N-tiles = 8 WMMA / K-step.
//    Double-buffered LDS, one barrier per K-step, prefetch 2 tiles ahead.
// ---------------------------------------------------------------------------
__global__ __launch_bounds__(256)
void gemm_proj_kernel(const float* __restrict__ W,   // [512,512]
                      const float* __restrict__ X,   // [B,512,49]
                      float* __restrict__ Y)         // [B,512,49]
{
    __shared__ __bf16 Wl[2][256][34];
    __shared__ __bf16 Xt[2][64][34];   // n-major (transposed) B tile
    const int b     = blockIdx.y;
    const int mBase = blockIdx.x * 256;
    const int t     = threadIdx.x;
    const int wave  = t >> 5, lane = t & 31;
    const int half  = lane >> 4, lm = lane & 15;

    v8f acc[2][4];
#pragma unroll
    for (int mi = 0; mi < 2; ++mi)
#pragma unroll
        for (int i = 0; i < 4; ++i)
#pragma unroll
            for (int e = 0; e < 8; ++e) acc[mi][i][e] = 0.f;

    float wreg[32];           // one full 32-wide W row per thread
    float xreg[8];

    const float* wrow = W + (size_t)(mBase + t) * CC;

    auto loadTiles = [&](int kk) {
        const float* src = wrow + kk;
#pragma unroll
        for (int i = 0; i < 32; ++i) wreg[i] = src[i];
#pragma unroll
        for (int i = 0; i < 8; ++i) {
            int l = t * 8 + i;
            int k = l & 31, n = l >> 5;
            xreg[i] = (n < NPIX) ? X[((size_t)b * CC + kk + k) * NPIX + n] : 0.f;
        }
    };
    auto storeTiles = [&](int buf) {
#pragma unroll
        for (int i = 0; i < 32; ++i) Wl[buf][t][i] = (__bf16)wreg[i];
#pragma unroll
        for (int i = 0; i < 8; ++i) {
            int l = t * 8 + i;
            Xt[buf][l >> 5][l & 31] = (__bf16)xreg[i];
        }
    };

    loadTiles(0);
    storeTiles(0);

    for (int it = 0; it < 16; ++it) {
        __syncthreads();
        if (it + 1 < 16) {
            loadTiles((it + 1) * 32);                       // overlap w/ WMMA
            if (it + 2 < 16) __builtin_prefetch(wrow + (it + 2) * 32, 0, 1);
        }
        const int buf = it & 1;
        v16bf af[2];
#pragma unroll
        for (int mi = 0; mi < 2; ++mi)
#pragma unroll
            for (int e = 0; e < 16; ++e)
                af[mi][e] = Wl[buf][wave * 32 + mi * 16 + lm][a_frag_k(e, half)];
#pragma unroll
        for (int nt = 0; nt < 4; ++nt) {
            v16bf bfr;
#pragma unroll
            for (int e = 0; e < 16; ++e) bfr[e] = Xt[buf][nt * 16 + lm][a_frag_k(e, half)];
#pragma unroll
            for (int mi = 0; mi < 2; ++mi)
                acc[mi][nt] = __builtin_amdgcn_wmma_f32_16x16x32_bf16(
                    false, af[mi], false, bfr, (short)0, acc[mi][nt], false, false);
        }
        if (it + 1 < 16) storeTiles((it + 1) & 1);
    }
#pragma unroll
    for (int mi = 0; mi < 2; ++mi)
#pragma unroll
        for (int nt = 0; nt < 4; ++nt) {
            int n = nt * 16 + lm;
            if (n >= NPIX) continue;
#pragma unroll
            for (int r = 0; r < 8; ++r) {
                int m = mBase + wave * 32 + mi * 16 + half * 8 + r;
                Y[((size_t)b * CC + m) * NPIX + n] = acc[mi][nt][r];
            }
        }
}

// ---------------------------------------------------------------------------
// 3) GroupNorm(32), channels contiguous per group, in place. grid (B, 32)
// ---------------------------------------------------------------------------
__global__ __launch_bounds__(256)
void groupnorm_kernel(float* __restrict__ X, const float* __restrict__ gamma,
                      const float* __restrict__ beta, int npix)
{
    int b = blockIdx.x, g = blockIdx.y;
    size_t base = ((size_t)b * CC + g * 16) * npix;
    int cnt = 16 * npix;
    float s = 0.f, sq = 0.f;
    for (int i = threadIdx.x; i < cnt; i += 256) {
        float v = X[base + i]; s += v; sq += v * v;
    }
    __shared__ float rs[256], rq[256];
    rs[threadIdx.x] = s; rq[threadIdx.x] = sq;
    __syncthreads();
    for (int o = 128; o > 0; o >>= 1) {
        if (threadIdx.x < o) {
            rs[threadIdx.x] += rs[threadIdx.x + o];
            rq[threadIdx.x] += rq[threadIdx.x + o];
        }
        __syncthreads();
    }
    float mean = rs[0] / cnt;
    float var  = rq[0] / cnt - mean * mean;
    float inv  = rsqrtf(var + EPSN);
    for (int i = threadIdx.x; i < cnt; i += 256) {
        int ch = g * 16 + i / npix;
        X[base + i] = (X[base + i] - mean) * inv * gamma[ch] + beta[ch];
    }
}

// ---------------------------------------------------------------------------
// 4) per-head depthwise 3x3 conv (groups=8) on vf -> raw v_local
// ---------------------------------------------------------------------------
__global__ __launch_bounds__(64)
void locconv_kernel(const float* __restrict__ VF,   // [B,512,49]
                    const float* __restrict__ W,    // [512,64,3,3]
                    float* __restrict__ Y)          // [B,512,49]
{
    int bo = blockIdx.x;                // b*512 + o
    int b = bo >> 9, o = bo & 511;
    int h = o >> 6;
    __shared__ float Vs[64][49];
    __shared__ float Ws[576];
    const float* src = VF + ((size_t)b * CC + h * 64) * NPIX;
    for (int i = threadIdx.x; i < 64 * NPIX; i += 64) Vs[i / NPIX][i % NPIX] = src[i];
    for (int i = threadIdx.x; i < 576; i += 64) Ws[i] = W[(size_t)o * 576 + i];
    __syncthreads();
    int n = threadIdx.x;
    if (n >= NPIX) return;
    int oy = n / 7, ox = n % 7;
    float acc = 0.f;
    for (int ci = 0; ci < 64; ++ci) {
#pragma unroll
        for (int r = 0; r < 3; ++r) {
            int iy = oy - 1 + r;
            if ((unsigned)iy >= 7u) continue;
#pragma unroll
            for (int s = 0; s < 3; ++s) {
                int ix = ox - 1 + s;
                if ((unsigned)ix >= 7u) continue;
                acc += Ws[ci * 9 + r * 3 + s] * Vs[ci][iy * 7 + ix];
            }
        }
    }
    Y[(size_t)bo * NPIX + n] = acc;
}

// ---------------------------------------------------------------------------
// 5) cosine-attention logits (+rel-pos bias). One block per (b,h).
//    q normalized over d (per token); k normalized over tokens (per d).
// ---------------------------------------------------------------------------
__global__ __launch_bounds__(256)
void qk_kernel(const float* __restrict__ QF, const float* __restrict__ KF,
               const float* __restrict__ logit_scale, // [8]
               const float* __restrict__ attn_bias,   // [8,49]
               const int*   __restrict__ bias_idx,    // [49,49]
               float* __restrict__ A)                 // [B,8,49,49]
{
    int bh = blockIdx.x;
    int b = bh >> 3, h = bh & 7;
    __shared__ float Q[64][49], K[64][49];
    __shared__ float qinv[49];
    const float* qs = QF + ((size_t)b * CC + h * 64) * NPIX;
    const float* ks = KF + ((size_t)b * CC + h * 64) * NPIX;
    for (int i = threadIdx.x; i < 64 * NPIX; i += 256) {
        Q[i / NPIX][i % NPIX] = qs[i];
        K[i / NPIX][i % NPIX] = ks[i];
    }
    __syncthreads();
    if (threadIdx.x < 64) {                       // normalize K rows (token axis)
        int dd = threadIdx.x;
        float s = 0.f;
        for (int n = 0; n < NPIX; ++n) s += K[dd][n] * K[dd][n];
        float inv = 1.f / fmaxf(sqrtf(s), 1e-12f);
        for (int n = 0; n < NPIX; ++n) K[dd][n] *= inv;
    } else if (threadIdx.x < 64 + NPIX) {         // q token norms (d axis)
        int n = threadIdx.x - 64;
        float s = 0.f;
        for (int dd = 0; dd < 64; ++dd) s += Q[dd][n] * Q[dd][n];
        qinv[n] = 1.f / fmaxf(sqrtf(s), 1e-12f);
    }
    __syncthreads();
    float ls = __expf(fminf(logit_scale[h], 4.6051702f));  // clamp at log(100)
    for (int idx = threadIdx.x; idx < NPIX * NPIX; idx += 256) {
        int nq = idx / NPIX, nk = idx % NPIX;
        float acc = 0.f;
        for (int dd = 0; dd < 64; ++dd) acc += Q[dd][nq] * K[dd][nk];
        acc = acc * qinv[nq] * ls + attn_bias[h * NPIX + bias_idx[idx]];
        A[(size_t)bh * (NPIX * NPIX) + idx] = acc;
    }
}

// ---------------------------------------------------------------------------
// 6) talking-head-1 -> softmax -> talking-head-2, per (b, nq) slice [8,49]
// ---------------------------------------------------------------------------
__global__ __launch_bounds__(64)
void th_softmax_kernel(float* __restrict__ A,
                       const float* __restrict__ th1,
                       const float* __restrict__ th2)
{
    int b = blockIdx.x, nq = blockIdx.y;
    __shared__ float a[8][49], s1[8][49];
    __shared__ float t1[64], t2[64];
    if (threadIdx.x < 64) { t1[threadIdx.x] = th1[threadIdx.x]; t2[threadIdx.x] = th2[threadIdx.x]; }
    for (int i = threadIdx.x; i < 8 * NPIX; i += 64) {
        int hh = i / NPIX, nk = i % NPIX;
        a[hh][nk] = A[(((size_t)b * 8 + hh) * NPIX + nq) * NPIX + nk];
    }
    __syncthreads();
    for (int i = threadIdx.x; i < 8 * NPIX; i += 64) {
        int ii = i / NPIX, nk = i % NPIX;
        float acc = 0.f;
#pragma unroll
        for (int j = 0; j < 8; ++j) acc += t1[ii * 8 + j] * a[j][nk];
        s1[ii][nk] = acc;
    }
    __syncthreads();
    if (threadIdx.x < 8) {
        int ii = threadIdx.x;
        float m = -1e30f;
        for (int nk = 0; nk < NPIX; ++nk) m = fmaxf(m, s1[ii][nk]);
        float ssum = 0.f;
        for (int nk = 0; nk < NPIX; ++nk) { float e = __expf(s1[ii][nk] - m); s1[ii][nk] = e; ssum += e; }
        float inv = 1.f / ssum;
        for (int nk = 0; nk < NPIX; ++nk) s1[ii][nk] *= inv;
    }
    __syncthreads();
    for (int i = threadIdx.x; i < 8 * NPIX; i += 64) {
        int ii = i / NPIX, nk = i % NPIX;
        float acc = 0.f;
#pragma unroll
        for (int j = 0; j < 8; ++j) acc += t2[ii * 8 + j] * s1[j][nk];
        A[(((size_t)b * 8 + ii) * NPIX + nq) * NPIX + nk] = acc;
    }
}

// ---------------------------------------------------------------------------
// 7) O = attn x V + v_local  -> o7 [B,512,49]. One block per (b,h).
// ---------------------------------------------------------------------------
__global__ __launch_bounds__(256)
void av_kernel(const float* __restrict__ A, const float* __restrict__ VF,
               const float* __restrict__ VLOC, float* __restrict__ O)
{
    int bh = blockIdx.x;
    int b = bh >> 3, h = bh & 7;
    __shared__ float As[49][49];
    __shared__ float Vs[64][49];
    const float* asrc = A + (size_t)bh * (NPIX * NPIX);
    for (int i = threadIdx.x; i < NPIX * NPIX; i += 256) As[i / NPIX][i % NPIX] = asrc[i];
    const float* vsrc = VF + ((size_t)b * CC + h * 64) * NPIX;
    for (int i = threadIdx.x; i < 64 * NPIX; i += 256) Vs[i / NPIX][i % NPIX] = vsrc[i];
    __syncthreads();
    for (int idx = threadIdx.x; idx < 64 * NPIX; idx += 256) {
        int dd = idx & 63, nq = idx >> 6;
        float acc = 0.f;
        for (int nk = 0; nk < NPIX; ++nk) acc += As[nq][nk] * Vs[dd][nk];
        size_t oidx = ((size_t)b * CC + h * 64 + dd) * NPIX + nq;
        O[oidx] = acc + VLOC[oidx];
    }
}

// ---------------------------------------------------------------------------
// 8) Fused: bilinear x8 upsample (half-pixel) + hardswish + WMMA GEMM out_w.
//    Y[b,m,p] = sum_c out_w[m,c] * hswish(bilerp(o7[b,c,7,7], p)), p in 56x56.
//    Double-buffered LDS; bilinear gathers hit the L2-resident 3.2MB o7.
// ---------------------------------------------------------------------------
__global__ __launch_bounds__(256)
void gemm_out_kernel(const float* __restrict__ W,   // [512,512]
                     const float* __restrict__ P,   // o7 [B,512,49]
                     float* __restrict__ Y)         // [B,512,3136]
{
    __shared__ __bf16 Wl[2][128][34];
    __shared__ __bf16 Xt[2][64][34];
    const int b     = blockIdx.z;
    const int mBase = blockIdx.x * 128;
    const int nBase = blockIdx.y * 64;
    const int t     = threadIdx.x;
    const int wave  = t >> 5, lane = t & 31;
    const int half  = lane >> 4, lm = lane & 15;

    // fixed output pixel handled by this thread's B-operand loader
    const int nloc = t >> 2;                   // 0..63
    const int p  = nBase + nloc;
    const int py = p / RR, px = p % RR;
    const float fy = (py + 0.5f) * 0.125f - 0.5f;
    const float fx = (px + 0.5f) * 0.125f - 0.5f;
    const int y0 = (int)floorf(fy), x0 = (int)floorf(fx);
    const float wy = fy - (float)y0, wx = fx - (float)x0;
    const int ya = min(max(y0, 0), 6),     yb = min(max(y0 + 1, 0), 6);
    const int xa = min(max(x0, 0), 6),     xb = min(max(x0 + 1, 0), 6);
    const int o00 = ya * 7 + xa, o01 = ya * 7 + xb;
    const int o10 = yb * 7 + xa, o11 = yb * 7 + xb;
    const float w00 = (1.f - wy) * (1.f - wx), w01 = (1.f - wy) * wx;
    const float w10 = wy * (1.f - wx),         w11 = wy * wx;
    const int kloc0 = (t & 3) * 8;

    v8f acc[4];
#pragma unroll
    for (int i = 0; i < 4; ++i)
#pragma unroll
        for (int e = 0; e < 8; ++e) acc[i][e] = 0.f;

    float wreg[16];
    float xg[8][4];
    const int   wrow  = t >> 1;
    const int   wkoff = (t & 1) * 16;
    const float* wsrc = W + (size_t)(mBase + wrow) * CC + wkoff;

    auto loadTiles = [&](int kk) {
#pragma unroll
        for (int i = 0; i < 16; ++i) wreg[i] = wsrc[kk + i];
#pragma unroll
        for (int i = 0; i < 8; ++i) {
            const float* g = P + ((size_t)b * CC + kk + kloc0 + i) * NPIX;
            xg[i][0] = g[o00]; xg[i][1] = g[o01];
            xg[i][2] = g[o10]; xg[i][3] = g[o11];
        }
    };
    auto storeTiles = [&](int buf) {
#pragma unroll
        for (int i = 0; i < 16; ++i) Wl[buf][wrow][wkoff + i] = (__bf16)wreg[i];
#pragma unroll
        for (int i = 0; i < 8; ++i) {
            float v = w00 * xg[i][0] + w01 * xg[i][1] + w10 * xg[i][2] + w11 * xg[i][3];
            v = v * fminf(fmaxf(v + 3.f, 0.f), 6.f) * (1.f / 6.f);   // hardswish
            Xt[buf][nloc][kloc0 + i] = (__bf16)v;
        }
    };

    loadTiles(0);
    storeTiles(0);

    for (int it = 0; it < 16; ++it) {
        __syncthreads();
        if (it + 1 < 16) {
            loadTiles((it + 1) * 32);                       // overlap w/ WMMA
            if (it + 2 < 16) __builtin_prefetch(wsrc + (it + 2) * 32, 0, 1);
        }
        const int buf = it & 1;
        v16bf af;
#pragma unroll
        for (int e = 0; e < 16; ++e)
            af[e] = Wl[buf][wave * 16 + lm][a_frag_k(e, half)];
#pragma unroll
        for (int nt = 0; nt < 4; ++nt) {
            v16bf bfr;
#pragma unroll
            for (int e = 0; e < 16; ++e) bfr[e] = Xt[buf][nt * 16 + lm][a_frag_k(e, half)];
            acc[nt] = __builtin_amdgcn_wmma_f32_16x16x32_bf16(
                false, af, false, bfr, (short)0, acc[nt], false, false);
        }
        if (it + 1 < 16) storeTiles((it + 1) & 1);
    }
#pragma unroll
    for (int nt = 0; nt < 4; ++nt) {
        int n = nBase + nt * 16 + lm;
#pragma unroll
        for (int r = 0; r < 8; ++r) {
            int m = mBase + wave * 16 + half * 8 + r;
            Y[((size_t)b * CC + m) * (RR * RR) + n] = acc[nt][r];
        }
    }
}

// ---------------------------------------------------------------------------
extern "C" void kernel_launch(void* const* d_in, const int* in_sizes, int n_in,
                              void* d_out, int out_size, void* d_ws, size_t ws_size,
                              hipStream_t stream) {
    (void)in_sizes; (void)n_in; (void)out_size; (void)ws_size;
    const float* query = (const float*)d_in[0];
    const float* key_i = (const float*)d_in[1];
    const float* value = (const float*)d_in[2];
    const float* qs_w = (const float*)d_in[3],  *qs_b = (const float*)d_in[4];
    const float* qs_g = (const float*)d_in[5],  *qs_bb = (const float*)d_in[6];
    const float* ks_w = (const float*)d_in[7],  *ks_b = (const float*)d_in[8];
    const float* ks_g = (const float*)d_in[9],  *ks_bb = (const float*)d_in[10];
    const float* vs_w = (const float*)d_in[11], *vs_b = (const float*)d_in[12];
    const float* vs_g = (const float*)d_in[13], *vs_bb = (const float*)d_in[14];
    const float* wq = (const float*)d_in[15], *gq_g = (const float*)d_in[16], *gq_b = (const float*)d_in[17];
    const float* wk = (const float*)d_in[18], *gk_g = (const float*)d_in[19], *gk_b = (const float*)d_in[20];
    const float* wv = (const float*)d_in[21], *gv_g = (const float*)d_in[22], *gv_b = (const float*)d_in[23];
    const float* loc_w = (const float*)d_in[24], *loc_g = (const float*)d_in[25], *loc_b = (const float*)d_in[26];
    const float* th1_w = (const float*)d_in[27], *th2_w = (const float*)d_in[28];
    const float* logit_scale = (const float*)d_in[29];
    const float* attn_bias   = (const float*)d_in[30];
    const int*   bias_idx    = (const int*)d_in[31];
    const float* out_w = (const float*)d_in[32], *out_g = (const float*)d_in[33], *out_b = (const float*)d_in[34];

    const size_t SL = (size_t)BB * CC * NPIX;     // 802816 floats
    float* ws   = (float*)d_ws;
    float* q4   = ws;
    float* k4   = q4 + SL;
    float* v4   = k4 + SL;
    float* qf   = v4 + SL;
    float* kf   = qf + SL;
    float* vf   = kf + SL;
    float* vloc = vf + SL;
    float* o7   = vloc + SL;
    float* attn = o7 + SL;                        // [32,8,49,49]

    // 1) strided depthwise conv + BN
    dwconv_bn_kernel<<<BB * CC, 64, 0, stream>>>(query, qs_w, qs_b, qs_g, qs_bb, q4);
    dwconv_bn_kernel<<<BB * CC, 64, 0, stream>>>(key_i, ks_w, ks_b, ks_g, ks_bb, k4);
    dwconv_bn_kernel<<<BB * CC, 64, 0, stream>>>(value, vs_w, vs_b, vs_g, vs_bb, v4);

    // 2) 1x1 projections (WMMA, 256 M rows/block) + GroupNorm
    gemm_proj_kernel<<<dim3(2, BB), 256, 0, stream>>>(wq, q4, qf);
    gemm_proj_kernel<<<dim3(2, BB), 256, 0, stream>>>(wk, k4, kf);
    gemm_proj_kernel<<<dim3(2, BB), 256, 0, stream>>>(wv, v4, vf);
    groupnorm_kernel<<<dim3(BB, 32), 256, 0, stream>>>(qf, gq_g, gq_b, NPIX);
    groupnorm_kernel<<<dim3(BB, 32), 256, 0, stream>>>(kf, gk_g, gk_b, NPIX);
    groupnorm_kernel<<<dim3(BB, 32), 256, 0, stream>>>(vf, gv_g, gv_b, NPIX);

    // 3) per-head local conv + GN -> v_local
    locconv_kernel<<<BB * CC, 64, 0, stream>>>(vf, loc_w, vloc);
    groupnorm_kernel<<<dim3(BB, 32), 256, 0, stream>>>(vloc, loc_g, loc_b, NPIX);

    // 4) cosine attention logits + rel-pos bias
    qk_kernel<<<BB * 8, 256, 0, stream>>>(qf, kf, logit_scale, attn_bias, bias_idx, attn);

    // 5) talking heads + softmax
    th_softmax_kernel<<<dim3(BB, NPIX), 64, 0, stream>>>(attn, th1_w, th2_w);

    // 6) attn x V + v_local
    av_kernel<<<BB * 8, 256, 0, stream>>>(attn, vf, vloc, o7);

    // 7) fused upsample + hardswish + output GEMM (WMMA) -> d_out (pre-GN)
    gemm_out_kernel<<<dim3(4, 49, BB), 256, 0, stream>>>(out_w, o7, (float*)d_out);

    // 8) GroupNorm in place on d_out
    groupnorm_kernel<<<dim3(BB, 32), 256, 0, stream>>>((float*)d_out, out_g, out_b, RR * RR);
}
